// RoleAwareMoE_64896955843140
// MI455X (gfx1250) — matrix operand order repaired and checked by reading, not compiled, not run on hardware
//
#include <hip/hip_runtime.h>
#include <hip/hip_bf16.h>
#include <math.h>

// ---------------------------------------------------------------------------
// Problem constants (match reference)
// ---------------------------------------------------------------------------
constexpr int C_DIM = 512;
constexpr int T_LEN = 4096;
constexpr int N_PROP = 8192;
constexpr int H_DIM = 1024;
constexpr int O_DIM = 512;

typedef __attribute__((ext_vector_type(16))) __bf16 v16bf;
typedef __attribute__((ext_vector_type(8)))  __bf16 v8bf;
typedef __attribute__((ext_vector_type(8)))  float  v8f;
typedef __attribute__((ext_vector_type(4)))  int    v4i;

// ---------------------------------------------------------------------------
// gfx1250 async global->LDS path (guarded; falls back to sync staging)
// ---------------------------------------------------------------------------
#if defined(__HIP_DEVICE_COMPILE__) && defined(__gfx1250__) && \
    __has_builtin(__builtin_amdgcn_global_load_async_to_lds_b128)
#define ASYNC_LDS 1
#else
#define ASYNC_LDS 0
#endif

__device__ __forceinline__ void async_copy16(const void* g, void* l) {
#if ASYNC_LDS
  __builtin_amdgcn_global_load_async_to_lds_b128(
      (__attribute__((address_space(1))) v4i*)const_cast<void*>(g),
      (__attribute__((address_space(3))) v4i*)l, 0, 0);
#endif
}

template <int N>
__device__ __forceinline__ void wait_async() {
#if ASYNC_LDS
#if __has_builtin(__builtin_amdgcn_s_wait_asynccnt)
  __builtin_amdgcn_s_wait_asynccnt(N);
#else
  asm volatile("s_wait_asynccnt %0" :: "i"(N) : "memory");
#endif
#endif
}

// ---------------------------------------------------------------------------
// Kernel 1: per-channel prefix sum of feat_map (C,T) -> csumT ((T+1),C)
// ---------------------------------------------------------------------------
__global__ void scan_kernel(const float* __restrict__ fm, float* __restrict__ csumT) {
  const int c = blockIdx.x;
  const int tid = threadIdx.x;
  __shared__ float buf[256];
  float carry = 0.0f;
  if (tid == 0) csumT[c] = 0.0f;
  for (int t0 = 0; t0 < T_LEN; t0 += 256) {
    float v = fm[(size_t)c * T_LEN + t0 + tid];
    buf[tid] = v;
    __syncthreads();
    #pragma unroll
    for (int off = 1; off < 256; off <<= 1) {
      float add = (tid >= off) ? buf[tid - off] : 0.0f;
      __syncthreads();
      buf[tid] += add;
      __syncthreads();
    }
    float inc = buf[tid] + carry;
    csumT[(size_t)(t0 + tid + 1) * C_DIM + c] = inc;
    carry += buf[255];
    __syncthreads();
  }
}

// ---------------------------------------------------------------------------
// Kernel 2: fused fp32 -> bf16 transpose of weights: W[K][N] -> Wt[N][K]
// ---------------------------------------------------------------------------
__global__ void transpose_bf16_kernel(const float* __restrict__ W, __bf16* __restrict__ Wt,
                                      int K, int Nc) {
  __shared__ float tile[32][33];
  const int k0 = blockIdx.x * 32, n0 = blockIdx.y * 32;
  const int tx = threadIdx.x & 31, ty = threadIdx.x >> 5;  // 32 x 8
  #pragma unroll
  for (int j = 0; j < 4; j++) {
    const int kk = ty + 8 * j;
    tile[kk][tx] = W[(unsigned)((k0 + kk) * Nc + n0 + tx)];
  }
  __syncthreads();
  #pragma unroll
  for (int j = 0; j < 4; j++) {
    const int nn = ty + 8 * j;
    Wt[(unsigned)((n0 + nn) * K + k0 + tx)] = (__bf16)tile[tx][nn];
  }
}

// ---------------------------------------------------------------------------
// Kernel 3: pooling via prefix-sum gathers + router MLP/softmax.
// ---------------------------------------------------------------------------
__global__ void pool_router_kernel(const float* __restrict__ csumT,
                                   const float* __restrict__ l_, const float* __restrict__ r_,
                                   const float* __restrict__ Wr1, const float* __restrict__ br1,
                                   const float* __restrict__ Wr2, const float* __restrict__ br2,
                                   __bf16* __restrict__ Xb, __bf16* __restrict__ Xi,
                                   __bf16* __restrict__ Xc,
                                   float* __restrict__ wts, float* __restrict__ out_w) {
  const int n = blockIdx.x;
  const int tid = threadIdx.x;
  const float l = l_[n];
  const float r = r_[n];
  const float w = fmaxf(r - l, 1.0f);
  const int rl = (int)rintf(l);
  const int rr = (int)rintf(r);
  const int bw = max(1, (int)floorf(0.15f * w));
  const int cw = max(2, (int)floorf(0.5f * w));
  const int li = max(0, rl);
  const int ri = min(T_LEN, rr + 1);

  int s[5], e[5];
  s[0] = max(0, rl - bw); e[0] = min(T_LEN, rl + bw);
  s[1] = max(0, rr - bw); e[1] = min(T_LEN, rr + bw);
  s[2] = li;              e[2] = max(li + 1, ri);
  s[3] = max(0, li - cw); e[3] = li;
  s[4] = ri;              e[4] = min(T_LEN, ri + cw);
  float inv[5];
  #pragma unroll
  for (int p = 0; p < 5; p++) {
    e[p] = min(max(s[p] + 1, e[p]), T_LEN);
    inv[p] = 1.0f / (float)(e[p] - s[p]);
  }

  const int c0 = tid * 2;
  #pragma unroll
  for (int p = 0; p < 5; p++) {
    const float2 hi = *(const float2*)&csumT[(unsigned)(e[p] * C_DIM + c0)];
    const float2 lo = *(const float2*)&csumT[(unsigned)(s[p] * C_DIM + c0)];
    const float v0 = (hi.x - lo.x) * inv[p];
    const float v1 = (hi.y - lo.y) * inv[p];
    __bf16* dst;
    switch (p) {
      case 0: dst = Xb + (size_t)n * (2 * C_DIM) + c0;            break;
      case 1: dst = Xb + (size_t)n * (2 * C_DIM) + C_DIM + c0;    break;
      case 2: dst = Xi + (size_t)n * C_DIM + c0;                  break;
      case 3: dst = Xc + (size_t)n * (2 * C_DIM) + c0;            break;
      default: dst = Xc + (size_t)n * (2 * C_DIM) + C_DIM + c0;   break;
    }
    dst[0] = (__bf16)v0;
    dst[1] = (__bf16)v1;
  }

  if (tid == 0) {
    const float Tf = (float)T_LEN;
    const float g[4] = { l / Tf, r / Tf, w / Tf, 0.5f * (l + r) / Tf };
    float h[32];
    #pragma unroll
    for (int j = 0; j < 32; j++) {
      float a = br1[j];
      #pragma unroll
      for (int k = 0; k < 4; k++) a += g[k] * Wr1[k * 32 + j];
      h[j] = fmaxf(a, 0.0f);
    }
    float lg[3];
    #pragma unroll
    for (int k = 0; k < 3; k++) {
      float a = br2[k];
      for (int j = 0; j < 32; j++) a += h[j] * Wr2[j * 3 + k];
      lg[k] = a;
    }
    float m = fmaxf(lg[0], fmaxf(lg[1], lg[2]));
    float e0 = __expf(lg[0] - m), e1 = __expf(lg[1] - m), e2 = __expf(lg[2] - m);
    float is = 1.0f / (e0 + e1 + e2);
    wts[n * 3 + 0] = e0 * is; out_w[n * 3 + 0] = e0 * is;
    wts[n * 3 + 1] = e1 * is; out_w[n * 3 + 1] = e1 * is;
    wts[n * 3 + 2] = e2 * is; out_w[n * 3 + 2] = e2 * is;
  }
}

// ---------------------------------------------------------------------------
// Kernel 4: bf16 WMMA GEMM, out = f(X[MxK] @ Wt[NxK]^T + bias)
// Block tile 128x64, BK=32, 8 waves, wave tile 32x32 (4 wmma / K-step).
// Double-buffered LDS; async global->LDS when available.
// MODE 0: ReLU -> bf16; MODE 1: scale*write f32; MODE 2: scale*accum f32.
// ---------------------------------------------------------------------------
#define LD8(p)  (*(const v8bf*)(p))
#define SHUF16(lo, hi) __builtin_shufflevector(lo, hi, 0,1,2,3,4,5,6,7,8,9,10,11,12,13,14,15)

template <int MODE>
__global__ __launch_bounds__(256)
void gemm_wmma_kernel(const __bf16* __restrict__ X, const __bf16* __restrict__ Wt,
                      const float* __restrict__ bias,
                      const float* __restrict__ wts, int expert,
                      void* __restrict__ outp, int M, int K, int Ncols) {
  constexpr int BM = 128, BN = 64, LDP = 40;       // 40 bf16 = 80B padded rows
  __shared__ __align__(16) __bf16 As[2][BM][LDP];
  __shared__ __align__(16) __bf16 Bs[2][BN][LDP];

  const int tid  = threadIdx.x;
  const int wave = tid >> 5;
  const int lane = tid & 31;
  const int lane15 = lane & 15;
  const int khalf  = lane >> 4;
  const int bm = blockIdx.x * BM;
  const int bn = blockIdx.y * BN;
  const int wm = (wave & 3) * 32;      // 0,32,64,96
  const int wn = (wave >> 2) * 32;     // 0,32

  // staging: A tile 128x32 -> 512 x 16B chunks (2/thread); B tile -> 256 (1/thread)
  const int aR0 = tid >> 2,          aC0 = (tid & 3) * 8;
  const int aR1 = (tid + 256) >> 2,  aC1 = (tid & 3) * 8;
  const int bR  = tid >> 2,          bC  = (tid & 3) * 8;

  const __bf16* Ab = X  + (unsigned)(bm * K);
  const __bf16* Bb = Wt + (unsigned)(bn * K);

  v8f acc00 = {}, acc01 = {}, acc10 = {}, acc11 = {};

  const int nk = K >> 5;

  auto stage_async = [&](int buf, int k0) {
    async_copy16(Ab + (unsigned)(aR0 * K + k0 + aC0), &As[buf][aR0][aC0]);
    async_copy16(Ab + (unsigned)(aR1 * K + k0 + aC1), &As[buf][aR1][aC1]);
    async_copy16(Bb + (unsigned)(bR  * K + k0 + bC ), &Bs[buf][bR ][bC ]);
  };

  auto compute = [&](int buf) {
    const int ra0 = wm + lane15, ra1 = wm + 16 + lane15;
    const int rb0 = wn + lane15, rb1 = wn + 16 + lane15;
    v16bf a0 = SHUF16(LD8(&As[buf][ra0][khalf * 8]), LD8(&As[buf][ra0][16 + khalf * 8]));
    v16bf a1 = SHUF16(LD8(&As[buf][ra1][khalf * 8]), LD8(&As[buf][ra1][16 + khalf * 8]));
    v16bf b0 = SHUF16(LD8(&Bs[buf][rb0][khalf * 8]), LD8(&Bs[buf][rb0][16 + khalf * 8]));
    v16bf b1 = SHUF16(LD8(&Bs[buf][rb1][khalf * 8]), LD8(&Bs[buf][rb1][16 + khalf * 8]));
    acc00 = __builtin_amdgcn_wmma_f32_16x16x32_bf16(false, a0, false, b0, (short)0, acc00, false, false);
    acc01 = __builtin_amdgcn_wmma_f32_16x16x32_bf16(false, a0, false, b1, (short)0, acc01, false, false);
    acc10 = __builtin_amdgcn_wmma_f32_16x16x32_bf16(false, a1, false, b0, (short)0, acc10, false, false);
    acc11 = __builtin_amdgcn_wmma_f32_16x16x32_bf16(false, a1, false, b1, (short)0, acc11, false, false);
  };

#if ASYNC_LDS
  stage_async(0, 0);
  for (int i = 0; i < nk; ++i) {
    const int buf = i & 1;
    if (i + 1 < nk) {
      stage_async(buf ^ 1, (i + 1) << 5);
      wait_async<3>();               // previous tile's 3 ops complete
    } else {
      wait_async<0>();
    }
    __syncthreads();
    compute(buf);
    __syncthreads();                 // protect buf^1 before next stage
  }
#else
  // synchronous fallback: register-staged double buffer
  float4 ra, rb, rc;
  auto gload = [&](int k0) {
    ra = *(const float4*)(Ab + (unsigned)(aR0 * K + k0 + aC0));
    rb = *(const float4*)(Ab + (unsigned)(aR1 * K + k0 + aC1));
    rc = *(const float4*)(Bb + (unsigned)(bR  * K + k0 + bC ));
  };
  auto lstore = [&](int buf) {
    *(float4*)&As[buf][aR0][aC0] = ra;
    *(float4*)&As[buf][aR1][aC1] = rb;
    *(float4*)&Bs[buf][bR ][bC ] = rc;
  };
  gload(0);
  lstore(0);
  __syncthreads();
  for (int i = 0; i < nk; ++i) {
    const int buf = i & 1;
    if (i + 1 < nk) gload((i + 1) << 5);
    compute(buf);
    __syncthreads();
    if (i + 1 < nk) { lstore(buf ^ 1); __syncthreads(); }
  }
#endif

  // epilogue: D layout: VGPR v -> M = v + 8*khalf, N = lane15
  const int mb = khalf * 8;
  #pragma unroll
  for (int mi = 0; mi < 2; mi++) {
    const v8f& aN0 = mi ? acc10 : acc00;
    const v8f& aN1 = mi ? acc11 : acc01;
    #pragma unroll
    for (int v = 0; v < 8; v++) {
      const int grow = bm + wm + mi * 16 + mb + v;
      const int gc0  = bn + wn + lane15;
      const int gc1  = gc0 + 16;
      float v0 = aN0[v] + bias[gc0];
      float v1 = aN1[v] + bias[gc1];
      if (MODE == 0) {
        __bf16* Hb = (__bf16*)outp;
        Hb[(unsigned)(grow * Ncols + gc0)] = (__bf16)fmaxf(v0, 0.0f);
        Hb[(unsigned)(grow * Ncols + gc1)] = (__bf16)fmaxf(v1, 0.0f);
      } else {
        const float s = wts[grow * 3 + expert];
        float* Of = (float*)outp;
        if (MODE == 1) {
          Of[(unsigned)(grow * Ncols + gc0)] = v0 * s;
          Of[(unsigned)(grow * Ncols + gc1)] = v1 * s;
        } else {
          Of[(unsigned)(grow * Ncols + gc0)] += v0 * s;
          Of[(unsigned)(grow * Ncols + gc1)] += v1 * s;
        }
      }
    }
  }
}

// ---------------------------------------------------------------------------
// Host launcher
// ---------------------------------------------------------------------------
extern "C" void kernel_launch(void* const* d_in, const int* in_sizes, int n_in,
                              void* d_out, int out_size, void* d_ws, size_t ws_size,
                              hipStream_t stream) {
  const float* feat = (const float*)d_in[0];
  const float* l    = (const float*)d_in[1];
  const float* r    = (const float*)d_in[2];
  const float* Wb1  = (const float*)d_in[3];
  const float* bb1  = (const float*)d_in[4];
  const float* Wb2  = (const float*)d_in[5];
  const float* bb2  = (const float*)d_in[6];
  const float* Wi1  = (const float*)d_in[7];
  const float* bi1  = (const float*)d_in[8];
  const float* Wi2  = (const float*)d_in[9];
  const float* bi2  = (const float*)d_in[10];
  const float* Wc1  = (const float*)d_in[11];
  const float* bc1  = (const float*)d_in[12];
  const float* Wc2  = (const float*)d_in[13];
  const float* bc2  = (const float*)d_in[14];
  const float* Wr1  = (const float*)d_in[15];
  const float* br1  = (const float*)d_in[16];
  const float* Wr2  = (const float*)d_in[17];
  const float* br2  = (const float*)d_in[18];

  char* p = (char*)d_ws;
  auto carve = [&](size_t bytes) {
    void* q = (void*)p;
    p += (bytes + 255) & ~(size_t)255;
    return q;
  };
  float*  csumT = (float*)carve((size_t)(T_LEN + 1) * C_DIM * 4);
  __bf16* Xb    = (__bf16*)carve((size_t)N_PROP * 2 * C_DIM * 2);
  __bf16* Xi    = (__bf16*)carve((size_t)N_PROP * C_DIM * 2);
  __bf16* Xc    = (__bf16*)carve((size_t)N_PROP * 2 * C_DIM * 2);
  __bf16* Hbuf  = (__bf16*)carve((size_t)N_PROP * H_DIM * 2);
  __bf16* tWb1  = (__bf16*)carve((size_t)2 * C_DIM * H_DIM * 2);
  __bf16* tWb2  = (__bf16*)carve((size_t)H_DIM * O_DIM * 2);
  __bf16* tWi1  = (__bf16*)carve((size_t)C_DIM * H_DIM * 2);
  __bf16* tWi2  = (__bf16*)carve((size_t)H_DIM * O_DIM * 2);
  __bf16* tWc1  = (__bf16*)carve((size_t)2 * C_DIM * H_DIM * 2);
  __bf16* tWc2  = (__bf16*)carve((size_t)H_DIM * O_DIM * 2);
  float*  wts   = (float*)carve((size_t)N_PROP * 3 * 4);

  float* out_enh = (float*)d_out;
  float* out_w   = out_enh + (size_t)N_PROP * O_DIM;

  // 1) prefix sums
  scan_kernel<<<C_DIM, 256, 0, stream>>>(feat, csumT);

  // 2) weight transpose + bf16 conversion: W[K][N] -> Wt[N][K]
  auto tr = [&](const float* W, __bf16* Wt, int K, int Nc) {
    transpose_bf16_kernel<<<dim3(K / 32, Nc / 32), 256, 0, stream>>>(W, Wt, K, Nc);
  };
  tr(Wb1, tWb1, 2 * C_DIM, H_DIM);
  tr(Wb2, tWb2, H_DIM, O_DIM);
  tr(Wi1, tWi1, C_DIM, H_DIM);
  tr(Wi2, tWi2, H_DIM, O_DIM);
  tr(Wc1, tWc1, 2 * C_DIM, H_DIM);
  tr(Wc2, tWc2, H_DIM, O_DIM);

  // 3) pooling + router
  pool_router_kernel<<<N_PROP, 256, 0, stream>>>(csumT, l, r, Wr1, br1, Wr2, br2,
                                                 Xb, Xi, Xc, wts, out_w);

  // 4) expert MLPs
  dim3 blk(256);
  dim3 g1h(N_PROP / 128, H_DIM / 64);   // (64,16)
  dim3 g1o(N_PROP / 128, O_DIM / 64);   // (64,8)

  gemm_wmma_kernel<0><<<g1h, blk, 0, stream>>>(Xb, tWb1, bb1, nullptr, 0, Hbuf,
                                               N_PROP, 2 * C_DIM, H_DIM);
  gemm_wmma_kernel<1><<<g1o, blk, 0, stream>>>(Hbuf, tWb2, bb2, wts, 0, out_enh,
                                               N_PROP, H_DIM, O_DIM);
  gemm_wmma_kernel<0><<<g1h, blk, 0, stream>>>(Xi, tWi1, bi1, nullptr, 0, Hbuf,
                                               N_PROP, C_DIM, H_DIM);
  gemm_wmma_kernel<2><<<g1o, blk, 0, stream>>>(Hbuf, tWi2, bi2, wts, 1, out_enh,
                                               N_PROP, H_DIM, O_DIM);
  gemm_wmma_kernel<0><<<g1h, blk, 0, stream>>>(Xc, tWc1, bc1, nullptr, 0, Hbuf,
                                               N_PROP, 2 * C_DIM, H_DIM);
  gemm_wmma_kernel<2><<<g1o, blk, 0, stream>>>(Hbuf, tWc2, bc2, wts, 2, out_enh,
                                               N_PROP, H_DIM, O_DIM);
}
